// LengthRegulator_44727789421050
// MI455X (gfx1250) — matrix operand order repaired and checked by reading, not compiled, not run on hardware
//
#include <hip/hip_runtime.h>

// ---------------------------------------------------------------------------
// FastSpeech duration-predictor + length-regulator for gfx1250 (MI455X).
// Convs lowered to bf16 WMMA (v_wmma_f32_16x16x32_bf16), f32 accumulate.
// ---------------------------------------------------------------------------

typedef __attribute__((ext_vector_type(16))) __bf16 v16bf;
typedef __attribute__((ext_vector_type(8)))  __bf16 v8bf;
typedef __attribute__((ext_vector_type(8)))  float  v8f;

#define B_    32
#define T_    512
#define D_    384      // channels (== F)
#define TOUT  4096
#define LNEPS 1e-5f

union BF16x16 { v16bf v; v8bf h[2]; };

// ---------------- prepack: fp32 -> bf16 elementwise ----------------
__global__ void k_cast_bf16(const float* __restrict__ src, __bf16* __restrict__ dst, int n) {
  int i = blockIdx.x * blockDim.x + threadIdx.x;
  if (i < n) dst[i] = (__bf16)src[i];
}

// ---------------- prepack conv weights [F,D,K] fp32 -> WMMA B-fragment order
// dst[(((k*12+ck)*24+nt)*32+lane)*16+e] = w[f = nt*16 + lane%16][c = ck*32 + (e/8)*16 + (lane/16)*8 + e%8][k]
// (ISA 7.12.2: 16-bit B 32x16, lane%16 = N, lane/16 selects K half, 16 bf16/lane)
__global__ void k_pack_w(const float* __restrict__ w, __bf16* __restrict__ dst) {
  int i = blockIdx.x * blockDim.x + threadIdx.x;
  if (i >= 3 * 12 * 24 * 32 * 16) return;
  int e    = i & 15;
  int lane = (i >> 4) & 31;
  int nt   = (i >> 9) % 24;
  int ck   = (i / (16 * 32 * 24)) % 12;
  int k    = i / (16 * 32 * 24 * 12);
  int f = nt * 16 + (lane & 15);
  int c = ck * 32 + (e >> 3) * 16 + (lane >> 4) * 8 + (e & 7);
  dst[i] = (__bf16)w[(f * D_ + c) * 3 + k];
}

// ---------------- fused conv1d(K=3,pad=1) + bias + LayerNorm + ReLU ----------
// Workgroup: 256 threads = 8 waves (2 M x 4 N). Tile: 64 time rows x 384 filters.
// Each wave: 2x6 accum tiles of 16x16 f32 (v8f each).
template <bool WRITE_DUR>
__global__ void __launch_bounds__(256) k_conv_ln(
    const __bf16* __restrict__ in,      // [B,T,384] bf16
    const __bf16* __restrict__ wfrag,   // packed weight fragments
    const float*  __restrict__ cbias,   // conv bias [384]
    const float*  __restrict__ ln_g,    // [384]
    const float*  __restrict__ ln_beta, // [384]
    const float*  __restrict__ lin_w,   // [384]
    const float*  __restrict__ lin_bias,// [1]
    __bf16*       __restrict__ out_h,   // [B,T,384] bf16 (WRITE_DUR=false)
    float*        __restrict__ out_dur) // [B,T]       (WRITE_DUR=true)
{
  extern __shared__ char smem[];
  __bf16* sX  = (__bf16*)smem;                                  // [66][384] bf16
  float*  sH  = (float*)(smem + 66 * D_ * 2);                   // [64][384] f32
  float*  sMu = (float*)(smem + 66 * D_ * 2 + 64 * D_ * 4);     // [64]
  float*  sRs = sMu + 64;                                       // [64]

  const int tid = threadIdx.x;
  const int b   = blockIdx.x >> 3;
  const int t0  = (blockIdx.x & 7) * 64;

  // Stage input rows t0-1 .. t0+64 (zero-padded at T boundaries), 128-bit chunks.
  const int ROWU4 = (D_ * 2) / 16; // 48 uint4 per row
  for (int i = tid; i < 66 * ROWU4; i += 256) {
    int r = i / ROWU4, q = i % ROWU4;
    int gr = t0 - 1 + r;
    uint4 v = make_uint4(0u, 0u, 0u, 0u);
    if (gr >= 0 && gr < T_)
      v = ((const uint4*)(in + ((size_t)b * T_ + gr) * D_))[q];
    ((uint4*)sX)[i] = v;
  }
  __syncthreads();

  const int lane = tid & 31;
  const int wave = tid >> 5;
  const int wm   = wave >> 2;   // 0..1 -> 32-row half
  const int wn   = wave & 3;    // 0..3 -> 96-filter chunk
  const int mrow = lane & 15;
  const int half = lane >> 4;

  v8f acc[2][6];
#pragma unroll
  for (int a = 0; a < 2; ++a)
#pragma unroll
    for (int c = 0; c < 6; ++c)
#pragma unroll
      for (int r = 0; r < 8; ++r) acc[a][c][r] = 0.f;

  for (int k = 0; k < 3; ++k) {
    for (int ck = 0; ck < 12; ++ck) {
      // A fragments (ISA 7.12.2 16-bit A 16x32): K = half*8 + e  /  16 + half*8 + e
      BF16x16 a0, a1;
      {
        const __bf16* base = sX + (wm * 32 + mrow + k) * D_ + ck * 32 + half * 8;
        a0.h[0] = *(const v8bf*)base;
        a0.h[1] = *(const v8bf*)(base + 16);
        base += 16 * D_;
        a1.h[0] = *(const v8bf*)base;
        a1.h[1] = *(const v8bf*)(base + 16);
      }
      const __bf16* wb =
          wfrag + ((((size_t)(k * 12 + ck) * 24 + wn * 6) * 32 + lane) * 16);
#pragma unroll
      for (int nt = 0; nt < 6; ++nt) {
        BF16x16 bf;
        bf.h[0] = *(const v8bf*)(wb + (size_t)nt * 512);
        bf.h[1] = *(const v8bf*)(wb + (size_t)nt * 512 + 8);
        acc[0][nt] = __builtin_amdgcn_wmma_f32_16x16x32_bf16(
            false, a0.v, false, bf.v, (short)0, acc[0][nt], false, false);
        acc[1][nt] = __builtin_amdgcn_wmma_f32_16x16x32_bf16(
            false, a1.v, false, bf.v, (short)0, acc[1][nt], false, false);
      }
    }
  }

  // Spill accumulators (+conv bias) to LDS in [row][col] f32 for LayerNorm.
  // C/D layout: VGPR r -> M = r + 8*half ; lane%16 -> N.
#pragma unroll
  for (int mt = 0; mt < 2; ++mt)
#pragma unroll
    for (int nt = 0; nt < 6; ++nt)
#pragma unroll
      for (int r = 0; r < 8; ++r) {
        int row = wm * 32 + mt * 16 + r + half * 8;
        int col = wn * 96 + nt * 16 + mrow;
        sH[row * D_ + col] = acc[mt][nt][r] + cbias[col];
      }
  __syncthreads();

  if (tid < 64) {
    const float4* p = (const float4*)(sH + tid * D_);
    float s = 0.f, s2 = 0.f;
#pragma unroll 4
    for (int q = 0; q < D_ / 4; ++q) {
      float4 v = p[q];
      s  += v.x + v.y + v.z + v.w;
      s2 += v.x * v.x + v.y * v.y + v.z * v.z + v.w * v.w;
    }
    float mu  = s * (1.f / D_);
    float var = s2 * (1.f / D_) - mu * mu;
    sMu[tid] = mu;
    sRs[tid] = rsqrtf(var + LNEPS);
  }
  __syncthreads();

  if (!WRITE_DUR) {
    for (int i = tid; i < 64 * D_; i += 256) {
      int row = i / D_, col = i % D_;
      float v = (sH[i] - sMu[row]) * sRs[row] * ln_g[col] + ln_beta[col];
      v = fmaxf(v, 0.f);
      out_h[((size_t)b * T_ + t0 + row) * D_ + col] = (__bf16)v;
    }
  } else {
    if (tid < 64) {
      int row = tid;
      float mu = sMu[row], rs = sRs[row], d = 0.f;
      for (int col = 0; col < D_; ++col) {
        float v = (sH[row * D_ + col] - mu) * rs * ln_g[col] + ln_beta[col];
        v = fmaxf(v, 0.f);
        d += v * lin_w[col];
      }
      out_dur[b * T_ + t0 + row] = d + lin_bias[0];
    }
  }
}

// ---------------- length regulator: scan + searchsorted(right) ----------------
__global__ void k_scan_idx(const int* __restrict__ target, int* __restrict__ idx) {
  __shared__ int s[T_];
  int b = blockIdx.x, t = threadIdx.x;
  s[t] = target[b * T_ + t];
  __syncthreads();
  for (int off = 1; off < T_; off <<= 1) {
    int v = (t >= off) ? s[t - off] : 0;
    __syncthreads();
    s[t] += v;
    __syncthreads();
  }
  for (int j = 0; j < TOUT / T_; ++j) {
    int m = t + j * T_;
    int lo = 0, hi = T_;
    while (lo < hi) {            // count of ends <= m
      int mid = (lo + hi) >> 1;
      if (s[mid] <= m) lo = mid + 1; else hi = mid;
    }
    idx[b * TOUT + m] = lo;
  }
}

// One block per (b, m) output row; 96 threads x float4 = 384 floats.
__global__ void __launch_bounds__(96) k_gather(const float* __restrict__ x,
                                               const int* __restrict__ idx,
                                               float* __restrict__ out) {
  int bm = blockIdx.x;
  int b  = bm >> 12;
  int id = idx[bm];
  bool valid = id < T_;
  int row = valid ? id : 0;
  float4 v = ((const float4*)(x + ((size_t)b * T_ + row) * D_))[threadIdx.x];
  if (!valid) v = make_float4(0.f, 0.f, 0.f, 0.f);
  ((float4*)(out + (size_t)bm * D_))[threadIdx.x] = v;
}

// ---------------------------------------------------------------------------
extern "C" void kernel_launch(void* const* d_in, const int* in_sizes, int n_in,
                              void* d_out, int out_size, void* d_ws, size_t ws_size,
                              hipStream_t stream) {
  (void)in_sizes; (void)n_in; (void)out_size; (void)ws_size;

  const float* x      = (const float*)d_in[0];
  const int*   target = (const int*)d_in[1];
  // d_in[2] = mel_max_length (== 4096, fixed by reference shapes / out_size)
  const float* c1w = (const float*)d_in[3];
  const float* c1b = (const float*)d_in[4];
  const float* g1  = (const float*)d_in[5];
  const float* b1  = (const float*)d_in[6];
  const float* c2w = (const float*)d_in[7];
  const float* c2b = (const float*)d_in[8];
  const float* g2  = (const float*)d_in[9];
  const float* b2  = (const float*)d_in[10];
  const float* lw  = (const float*)d_in[11];
  const float* lb  = (const float*)d_in[12];

  char* ws = (char*)d_ws;
  const size_t NW = (size_t)3 * 12 * 24 * 32 * 16;           // packed weight elems
  __bf16* xb  = (__bf16*)ws;                 size_t off = (size_t)B_ * T_ * D_ * 2;
  __bf16* wf1 = (__bf16*)(ws + off);         off += NW * 2;
  __bf16* wf2 = (__bf16*)(ws + off);         off += NW * 2;
  __bf16* h1  = (__bf16*)(ws + off);         off += (size_t)B_ * T_ * D_ * 2;
  int*    idx = (int*)(ws + off);            // [B,TOUT]

  float* out = (float*)d_out;                       // [B,TOUT,D]
  float* dur = out + (size_t)B_ * TOUT * D_;        // [B,T]

  int nx = B_ * T_ * D_;
  k_cast_bf16<<<(nx + 255) / 256, 256, 0, stream>>>(x, xb, nx);
  k_pack_w<<<((int)NW + 255) / 256, 256, 0, stream>>>(c1w, wf1);
  k_pack_w<<<((int)NW + 255) / 256, 256, 0, stream>>>(c2w, wf2);

  size_t smem = 66 * D_ * 2 + 64 * D_ * 4 + 128 * 4;  // ~149.5 KB of 320 KB WGP LDS
  k_conv_ln<false><<<B_ * (T_ / 64), 256, smem, stream>>>(
      xb, wf1, c1b, g1, b1, lw, lb, h1, nullptr);
  k_conv_ln<true><<<B_ * (T_ / 64), 256, smem, stream>>>(
      h1, wf2, c2b, g2, b2, lw, lb, nullptr, dur);

  k_scan_idx<<<B_, T_, 0, stream>>>(target, idx);
  k_gather<<<B_ * TOUT, 96, 0, stream>>>(x, idx, out);
}
